// MultiAttention_30511447670883
// MI455X (gfx1250) — matrix operand order repaired and checked by reading, not compiled, not run on hardware
//
#include <hip/hip_runtime.h>

// ---------------------------------------------------------------------------
// Fused transformer block for MI455X (gfx1250, wave32, WMMA).
//
// Roofline: 193 GFLOP vs ~270 MB mandatory HBM traffic (11.6 us @ 23.3 TB/s)
// -> compute (f16 WMMA) bound when fully fused.  One block = 32 batch
// elements; all intermediates live in LDS (~245 KB of the 320 KB WGP LDS).
// All GEMMs use v_wmma_f32_16x16x32_f16 with 2x2 register blocking
// (4 accumulators per wave: 1 fragment load per WMMA, independent chains).
// LayerNorm affine terms are folded into the f16 weights at prep time:
//   LN(x)@W = rs*(x .* g)@W - mu*rs*(g@W) + be@W
// so only per-row {mu, rs} stats are computed on-chip (parallel ds_add_f32
// reductions), LN1 needs no materialization and LN2 is an in-place row scale.
// ---------------------------------------------------------------------------

typedef _Float16 h16;
typedef __attribute__((ext_vector_type(16))) _Float16 v16h;
typedef __attribute__((ext_vector_type(8)))  _Float16 v8h;
typedef __attribute__((ext_vector_type(8)))  float    v8f;

union V16U { v16h v; struct { v8h lo, hi; } p; };

#define MSTR 264   // LDS row stride (halfs) for 256-wide buffers (528B rows, 16B-aligned)
#define QSTR 104   // LDS row stride (halfs) for per-head 96-wide qkv buffer

// A fragment (16x32 f16, MxK) from LDS row-major.
// ISA 7.12.2: lane<16 -> M=lane, K in {b..b+7} U {b+16..b+23}, b=(lane>>4)*8.
__device__ __forceinline__ v16h frag_a_lds(const h16* base, int row0, int ld,
                                           int k0, int lane) {
  int m    = lane & 15;
  int ksel = (lane >> 4) << 3;
  const h16* p = base + (size_t)(row0 + m) * ld + k0 + ksel;
  V16U u;
  u.p.lo = *(const v8h*)p;
  u.p.hi = *(const v8h*)(p + 16);
  return u.v;
}

// B fragment (32x16 f16, KxN) from global, stored as B^T (N-major, ld = K).
// B striping matches A: lane<16 -> N=lane, K in {b..b+7} U {b+16..b+23}, b=(lane>>4)*8.
__device__ __forceinline__ v16h frag_b_glb(const h16* Wt, int n0, int ldk,
                                           int k0, int lane) {
  int n    = n0 + (lane & 15);
  int ksel = (lane >> 4) << 3;
  const h16* p = Wt + (size_t)n * ldk + k0 + ksel;
  V16U u;
  u.p.lo = *(const v8h*)p;
  u.p.hi = *(const v8h*)(p + 16);
  return u.v;
}

#define WMMA(acc, a, b) \
  acc = __builtin_amdgcn_wmma_f32_16x16x32_f16(false, a, false, b, (short)0, acc, false, false)

// ---------------------------------------------------------------------------
// Prep kernel: f16 B^T weights (+LN gamma folded in) and LN correction vectors.
//   wsH[0      .. 49152)  WqkvT [n=head*96+j][k], ld 128
//   wsH[49152  .. 114688) WfT'  [n][k] = g1[k]*Wf[k][n], ld 256
//   wsH[114688 .. 901120) WsT'  [n][kk]= g2[kk%512]*Ws[kk][n], ld 3072
//   wsF[0..256)    cg  = g1 @ Wf
//   wsF[256..512)  cb  = be1 @ Wf + bf
//   wsF[512..2048) cgs[s][n] = g2 @ Ws_s
//   wsF[2048..2304) cbs = sum_s be2 @ Ws_s + bs
// ---------------------------------------------------------------------------
__global__ void prep_weights(const float* __restrict__ W0, const float* __restrict__ W1,
                             const float* __restrict__ W2, const float* __restrict__ W3,
                             const float* __restrict__ Wf, const float* __restrict__ Ws,
                             const float* __restrict__ g1, const float* __restrict__ be1,
                             const float* __restrict__ bf,
                             const float* __restrict__ g2, const float* __restrict__ be2,
                             const float* __restrict__ bs,
                             h16* __restrict__ wsH, float* __restrict__ wsF) {
  int i = blockIdx.x * blockDim.x + threadIdx.x;
  if (i < 384 * 128) {
    int n = i >> 7, k = i & 127;
    const float* W = (n < 96) ? W0 : (n < 192) ? W1 : (n < 288) ? W2 : W3;
    wsH[i] = (h16)W[k * 96 + (n % 96)];
    return;
  }
  i -= 384 * 128;
  if (i < 256 * 256) {
    int n = i >> 8, k = i & 255;
    wsH[49152 + i] = (h16)(g1[k] * Wf[k * 256 + n]);
    return;
  }
  i -= 256 * 256;
  if (i < 256 * 3072) {
    int n = i / 3072, k = i - n * 3072;
    wsH[114688 + i] = (h16)(g2[k & 511] * Ws[(size_t)k * 256 + n]);
    return;
  }
  i -= 256 * 3072;
  if (i < 256) {
    float a = 0.f;
    for (int k = 0; k < 256; ++k) a += g1[k] * Wf[k * 256 + i];
    wsF[i] = a;
    return;
  }
  i -= 256;
  if (i < 256) {
    float a = bf[i];
    for (int k = 0; k < 256; ++k) a += be1[k] * Wf[k * 256 + i];
    wsF[256 + i] = a;
    return;
  }
  i -= 256;
  if (i < 1536) {
    int s = i >> 8, n = i & 255;
    float a = 0.f;
    for (int k = 0; k < 512; ++k) a += g2[k] * Ws[(size_t)(s * 512 + k) * 256 + n];
    wsF[512 + i] = a;
    return;
  }
  i -= 1536;
  if (i < 256) {
    float a = bs[i];
    for (int kk = 0; kk < 3072; ++kk) a += be2[kk & 511] * Ws[(size_t)kk * 256 + i];
    wsF[2048 + i] = a;
  }
}

// ---------------------------------------------------------------------------
// Fused kernel: 256 threads (8 waves), 32 batch / block, 2048 blocks.
// LDS rows are s-major: row r = s*32 + b  (s 0..5, b 0..31) so the final
// K=3072 GEMM splits into 12 clean 16-row WMMA tiles (2 batch halves per s).
// ---------------------------------------------------------------------------
__global__ __launch_bounds__(256, 1)
void fused_block(const float* __restrict__ x,
                 const float* __restrict__ b0, const float* __restrict__ b1,
                 const float* __restrict__ b2, const float* __restrict__ b3,
                 const h16* __restrict__ WqkvT,
                 const h16* __restrict__ WfT,
                 const h16* __restrict__ WsT,
                 const float* __restrict__ cg,  const float* __restrict__ cb,
                 const float* __restrict__ cgs, const float* __restrict__ cbs,
                 float* __restrict__ out) {
  __shared__ __align__(16) h16 sM [192 * MSTR];  // m = [x | attn]; LN2-scaled in place
  __shared__ __align__(16) h16 sN2[192 * MSTR];  // n2 = LN1(m)@Wf'; LN2-scaled in place
  __shared__ __align__(16) h16 sQ [192 * QSTR];  // per-head qkv
  __shared__ float sScores[32 * 36];
  __shared__ float sSum[192], sSq[192];
  __shared__ float sAl[192], sBe[192];           // LN1: rs, -mu*rs ; LN2: rs reuses sAl
  __shared__ float sBeta2[192];                  // LN2: -mu*rs

  const int tid  = threadIdx.x;
  const int wave = tid >> 5;
  const int lane = tid & 31;
  const int cl   = lane & 15;
  const int hi   = (lane >> 4) << 3;
  const int bat0 = blockIdx.x << 5;              // 2048 blocks * 32 batch

  // ---- Stage 0: x (32x6x128 f32) -> f16 LDS, rows r = s*32 + b ----
  for (int e = tid; e < 192 * 16; e += 256) {
    int gr = e >> 4;                             // b*6+s
    int c8 = (e & 15) << 3;
    int bb = gr / 6, ss = gr - bb * 6;
    const float* src = x + (((size_t)(bat0 + bb)) * 6 + ss) * 128 + c8;
    float4 v0 = *(const float4*)src;
    float4 v1 = *(const float4*)(src + 4);
    v8h r;
    r[0] = (h16)v0.x; r[1] = (h16)v0.y; r[2] = (h16)v0.z; r[3] = (h16)v0.w;
    r[4] = (h16)v1.x; r[5] = (h16)v1.y; r[6] = (h16)v1.z; r[7] = (h16)v1.w;
    *(v8h*)(sM + (ss * 32 + bb) * MSTR + c8) = r;
  }
  __syncthreads();

  // ---- Stage 1: per head, QKV GEMM (192x96x128, 2x2 blocked) + attention ----
  for (int hh = 0; hh < 4; ++hh) {
    const float* bh = (hh == 0) ? b0 : (hh == 1) ? b1 : (hh == 2) ? b2 : b3;
    for (int t = wave; t < 18; t += 8) {         // 6 row-pairs x 3 col-pairs
      int sp = t / 3, np = t - sp * 3;
      int row0 = sp * 32, row1 = row0 + 16;
      v8f a00 = {}, a01 = {}, a10 = {}, a11 = {};
#pragma unroll
      for (int k0 = 0; k0 < 128; k0 += 32) {
        v16h fa0 = frag_a_lds(sM, row0, MSTR, k0, lane);
        v16h fa1 = frag_a_lds(sM, row1, MSTR, k0, lane);
        v16h fb0 = frag_b_glb(WqkvT, hh * 96 + np * 32,      128, k0, lane);
        v16h fb1 = frag_b_glb(WqkvT, hh * 96 + np * 32 + 16, 128, k0, lane);
        WMMA(a00, fa0, fb0); WMMA(a01, fa0, fb1);
        WMMA(a10, fa1, fb0); WMMA(a11, fa1, fb1);
      }
      int c0 = np * 32 + cl, c1 = c0 + 16;       // head-local qkv columns
      float bi0 = bh[c0], bi1 = bh[c1];
#pragma unroll
      for (int r = 0; r < 8; ++r) {
        int r0 = row0 + hi + r, r1 = row1 + hi + r;
        sQ[r0 * QSTR + c0] = (h16)(a00[r] + bi0);
        sQ[r0 * QSTR + c1] = (h16)(a01[r] + bi1);
        sQ[r1 * QSTR + c0] = (h16)(a10[r] + bi0);
        sQ[r1 * QSTR + c1] = (h16)(a11[r] + bi1);
      }
    }
    __syncthreads();
    // scores[b][q][k] = Q(q).K(k)  (reference has no softmax / no scaling)
    for (int e = tid; e < 1152; e += 256) {
      int bb = e / 36, rem = e - bb * 36;
      int q = rem / 6, k = rem - q * 6;
      const h16* Qp = sQ + (q * 32 + bb) * QSTR;
      const h16* Kp = sQ + (k * 32 + bb) * QSTR + 32;
      float acc = 0.f;
#pragma unroll
      for (int d = 0; d < 32; ++d) acc += (float)Qp[d] * (float)Kp[d];
      sScores[bb * 36 + rem] = acc;
    }
    __syncthreads();
    // attn[b][q][c] = sum_k scores * V(k)[c]  -> sM cols 128 + 32*hh
    for (int e = tid; e < 6144; e += 256) {
      int c = e & 31, rem = e >> 5;
      int bb = rem / 6, q = rem - bb * 6;
      const float* sc = sScores + bb * 36 + q * 6;
      float acc = 0.f;
#pragma unroll
      for (int k = 0; k < 6; ++k)
        acc += sc[k] * (float)sQ[(k * 32 + bb) * QSTR + 64 + c];
      sM[(q * 32 + bb) * MSTR + 128 + hh * 32 + c] = (h16)acc;
    }
    __syncthreads();
  }

  // ---- Stage 2: LN1 stats over m(256): parallel ds_add_f32 reduction ----
  if (tid < 192) { sSum[tid] = 0.f; sSq[tid] = 0.f; }
  __syncthreads();
  for (int e = tid; e < 192 * 32; e += 256) {
    int row = e % 192;                           // lanes -> distinct rows
    int c0 = (e / 192) << 3;
    v8h v = *(const v8h*)(sM + row * MSTR + c0);
    float s = 0.f, s2 = 0.f;
#pragma unroll
    for (int i = 0; i < 8; ++i) { float f = (float)v[i]; s += f; s2 += f * f; }
    atomicAdd(&sSum[row], s);
    atomicAdd(&sSq[row], s2);
  }
  __syncthreads();
  if (tid < 192) {
    float mu = sSum[tid] * (1.f / 256.f);
    float rs = rsqrtf(sSq[tid] * (1.f / 256.f) - mu * mu + 1e-5f);
    sAl[tid] = rs; sBe[tid] = -mu * rs;
    sSum[tid] = 0.f; sSq[tid] = 0.f;             // re-zero for LN2
  }
  __syncthreads();

  // ---- Stage 3: n2 = rs*(m @ Wf') + (-mu*rs)*cg + cb   (192x256x256) ----
  for (int t = wave; t < 48; t += 8) {           // 6 row-pairs x 8 col-pairs
    int sp = t >> 3, np = t & 7;
    int row0 = sp * 32, row1 = row0 + 16;
    v8f a00 = {}, a01 = {}, a10 = {}, a11 = {};
#pragma unroll
    for (int k0 = 0; k0 < 256; k0 += 32) {
      v16h fa0 = frag_a_lds(sM, row0, MSTR, k0, lane);
      v16h fa1 = frag_a_lds(sM, row1, MSTR, k0, lane);
      v16h fb0 = frag_b_glb(WfT, np * 32,      256, k0, lane);
      v16h fb1 = frag_b_glb(WfT, np * 32 + 16, 256, k0, lane);
      WMMA(a00, fa0, fb0); WMMA(a01, fa0, fb1);
      WMMA(a10, fa1, fb0); WMMA(a11, fa1, fb1);
    }
    int c0 = np * 32 + cl, c1 = c0 + 16;
    float cg0 = cg[c0], cb0 = cb[c0], cg1 = cg[c1], cb1 = cb[c1];
#pragma unroll
    for (int r = 0; r < 8; ++r) {
      int r0 = row0 + hi + r, r1 = row1 + hi + r;
      float A0 = sAl[r0], B0 = sBe[r0], A1 = sAl[r1], B1 = sBe[r1];
      sN2[r0 * MSTR + c0] = (h16)(A0 * a00[r] + B0 * cg0 + cb0);
      sN2[r0 * MSTR + c1] = (h16)(A0 * a01[r] + B0 * cg1 + cb1);
      sN2[r1 * MSTR + c0] = (h16)(A1 * a10[r] + B1 * cg0 + cb0);
      sN2[r1 * MSTR + c1] = (h16)(A1 * a11[r] + B1 * cg1 + cb1);
    }
  }
  __syncthreads();

  // ---- Stage 4: LN2 stats over f=[n2|m] (512), then in-place scale by rs ----
  for (int e = tid; e < 192 * 64; e += 256) {
    int row = e % 192;
    int cc = e / 192;                            // 0..63
    const h16* p = (cc < 32) ? (sN2 + row * MSTR + (cc << 3))
                             : (sM  + row * MSTR + ((cc - 32) << 3));
    v8h v = *(const v8h*)p;
    float s = 0.f, s2 = 0.f;
#pragma unroll
    for (int i = 0; i < 8; ++i) { float f = (float)v[i]; s += f; s2 += f * f; }
    atomicAdd(&sSum[row], s);
    atomicAdd(&sSq[row], s2);
  }
  __syncthreads();
  if (tid < 192) {
    float mu = sSum[tid] * (1.f / 512.f);
    float rs = rsqrtf(sSq[tid] * (1.f / 512.f) - mu * mu + 1e-5f);
    sAl[tid] = rs;                               // rs2
    sBeta2[tid] = -mu * rs;
  }
  __syncthreads();
  for (int e = tid; e < 192 * 64; e += 256) {
    int row = e % 192;
    int cc = e / 192;
    h16* p = (cc < 32) ? (sN2 + row * MSTR + (cc << 3))
                       : (sM  + row * MSTR + ((cc - 32) << 3));
    float rs = sAl[row];
    v8h v = *(const v8h*)p, o;
#pragma unroll
    for (int i = 0; i < 8; ++i) o[i] = (h16)((float)v[i] * rs);
    *(v8h*)p = o;
  }
  __syncthreads();

  // ---- Stage 5: out = relu(sum_s fhat_s @ Ws'_s + beta2.cgs + cbs) ----
  {
    const int nt0 = wave, nt1 = wave + 8;
    v8f a00 = {}, a01 = {}, a10 = {}, a11 = {};  // [batch-half][nt]
    for (int s = 0; s < 6; ++s) {
      int rb = s * 32;
#pragma unroll
      for (int k0 = 0; k0 < 512; k0 += 32) {
        v16h fa0 = (k0 < 256) ? frag_a_lds(sN2, rb,      MSTR, k0,       lane)
                              : frag_a_lds(sM,  rb,      MSTR, k0 - 256, lane);
        v16h fa1 = (k0 < 256) ? frag_a_lds(sN2, rb + 16, MSTR, k0,       lane)
                              : frag_a_lds(sM,  rb + 16, MSTR, k0 - 256, lane);
        v16h fb0 = frag_b_glb(WsT, nt0 * 16, 3072, s * 512 + k0, lane);
        v16h fb1 = frag_b_glb(WsT, nt1 * 16, 3072, s * 512 + k0, lane);
        WMMA(a00, fa0, fb0); WMMA(a01, fa0, fb1);
        WMMA(a10, fa1, fb0); WMMA(a11, fa1, fb1);
      }
    }
    int c0 = nt0 * 16 + cl, c1 = nt1 * 16 + cl;
    float cgs0[6], cgs1[6];
#pragma unroll
    for (int s = 0; s < 6; ++s) { cgs0[s] = cgs[s * 256 + c0]; cgs1[s] = cgs[s * 256 + c1]; }
    float cb0 = cbs[c0], cb1 = cbs[c1];
#pragma unroll
    for (int r = 0; r < 8; ++r) {
      int bl0 = hi + r, bl1 = 16 + hi + r;       // local batch indices
      float v00 = a00[r] + cb0, v01 = a01[r] + cb1;
      float v10 = a10[r] + cb0, v11 = a11[r] + cb1;
#pragma unroll
      for (int s = 0; s < 6; ++s) {
        float b2a = sBeta2[s * 32 + bl0];
        float b2b = sBeta2[s * 32 + bl1];
        v00 += b2a * cgs0[s]; v01 += b2a * cgs1[s];
        v10 += b2b * cgs0[s]; v11 += b2b * cgs1[s];
      }
      size_t ob0 = (size_t)(bat0 + bl0) * 256, ob1 = (size_t)(bat0 + bl1) * 256;
      out[ob0 + c0] = v00 > 0.f ? v00 : 0.f;
      out[ob0 + c1] = v01 > 0.f ? v01 : 0.f;
      out[ob1 + c0] = v10 > 0.f ? v10 : 0.f;
      out[ob1 + c1] = v11 > 0.f ? v11 : 0.f;
    }
  }
}

extern "C" void kernel_launch(void* const* d_in, const int* in_sizes, int n_in,
                              void* d_out, int out_size, void* d_ws, size_t ws_size,
                              hipStream_t stream) {
  const float* x   = (const float*)d_in[0];
  const float* W0  = (const float*)d_in[1];
  const float* b0  = (const float*)d_in[2];
  const float* W1  = (const float*)d_in[3];
  const float* b1  = (const float*)d_in[4];
  const float* W2  = (const float*)d_in[5];
  const float* b2  = (const float*)d_in[6];
  const float* W3  = (const float*)d_in[7];
  const float* b3  = (const float*)d_in[8];
  const float* g1  = (const float*)d_in[9];
  const float* be1 = (const float*)d_in[10];
  const float* Wf  = (const float*)d_in[11];
  const float* bf  = (const float*)d_in[12];
  const float* g2  = (const float*)d_in[13];
  const float* be2 = (const float*)d_in[14];
  const float* Ws  = (const float*)d_in[15];
  const float* bs  = (const float*)d_in[16];

  h16*   wsH = (h16*)d_ws;
  float* wsF = (float*)(wsH + 901120);
  const h16* WqkvT = wsH;
  const h16* WfT   = wsH + 49152;
  const h16* WsT   = wsH + 114688;

  // 49152 + 65536 + 786432 + 256 + 256 + 1536 + 256 = 903424 = 3529 * 256
  prep_weights<<<3529, 256, 0, stream>>>(W0, W1, W2, W3, Wf, Ws,
                                         g1, be1, bf, g2, be2, bs, wsH, wsF);
  fused_block<<<65536 / 32, 256, 0, stream>>>(x, b0, b1, b2, b3,
                                              WqkvT, WfT, WsT,
                                              wsF, wsF + 256, wsF + 512, wsF + 2048,
                                              (float*)d_out);
}
